// RingAttention_66022237274182
// MI455X (gfx1250) — matrix operand order repaired and checked by reading, not compiled, not run on hardware
//
#include <hip/hip_runtime.h>
#include <hip/hip_bf16.h>
#include <math.h>

typedef _Float16 half_t;
typedef __attribute__((ext_vector_type(16))) _Float16 v16h;
typedef __attribute__((ext_vector_type(8)))  _Float16 v8h;
typedef __attribute__((ext_vector_type(4)))  _Float16 v4h;
typedef __attribute__((ext_vector_type(8)))  float    v8f;
typedef int v4i __attribute__((vector_size(16)));   // matches builtin param type

#define BATCH  2
#define SEQ    4096
#define DMODEL 1024
#define HEADS  16
#define DHEAD  64
#define BSROWS (BATCH * SEQ)   // 8192

#if __has_builtin(__builtin_amdgcn_global_load_async_to_lds_b128) && \
    __has_builtin(__builtin_amdgcn_s_wait_asynccnt)
#define USE_ASYNC_LDS 1
#else
#define USE_ASYNC_LDS 0
#endif

typedef __attribute__((address_space(1))) v4i* gptr_v4i;
typedef __attribute__((address_space(3))) v4i* lptr_v4i;

__device__ __forceinline__ v8f wmma_f16(v16h a, v16h b, v8f c) {
  return __builtin_amdgcn_wmma_f32_16x16x32_f16(
      false, a, false, b, (short)0, c, false, false);
}

union frag16 { v16h v; v8h h[2]; };

// ---------------------------------------------------------------------------
// Repack 1: f32 -> f16 elementwise, 4 elems/thread
// ---------------------------------------------------------------------------
__global__ __launch_bounds__(256) void cvt_f32_to_f16(
    const float* __restrict__ in, half_t* __restrict__ out) {
  const int t = (blockIdx.x * 256 + threadIdx.x) * 4;
  float4 f = *(const float4*)(in + t);
  v4h h;
  h[0] = (half_t)f.x; h[1] = (half_t)f.y; h[2] = (half_t)f.z; h[3] = (half_t)f.w;
  *(v4h*)(out + t) = h;
}

// ---------------------------------------------------------------------------
// Repack 2: W[K][N] f32 -> Wt[N][K] f16 via LDS 32x32 tile (both sides coalesced)
// gridDim.z selects the weight (0..3 -> wq,wk,wv,wo)
// ---------------------------------------------------------------------------
__global__ __launch_bounds__(256) void cvt_transpose_w(
    const float* __restrict__ wq, const float* __restrict__ wk,
    const float* __restrict__ wv, const float* __restrict__ wo,
    half_t* __restrict__ wt) {
  __shared__ float tile[32][33];
  const float* w = (blockIdx.z == 0) ? wq : (blockIdx.z == 1) ? wk
                  : (blockIdx.z == 2) ? wv : wo;
  half_t* dst = wt + (size_t)blockIdx.z * DMODEL * DMODEL;
  const int k0 = blockIdx.x * 32, n0 = blockIdx.y * 32;
  const int tx = threadIdx.x & 31, ty = threadIdx.x >> 5;   // 32 x 8
#pragma unroll
  for (int i = 0; i < 32; i += 8)
    tile[ty + i][tx] = w[(size_t)(k0 + ty + i) * DMODEL + n0 + tx];
  __syncthreads();
#pragma unroll
  for (int i = 0; i < 32; i += 8)
    dst[(size_t)(n0 + ty + i) * DMODEL + k0 + tx] = (half_t)tile[tx][ty + i];
}

// ---------------------------------------------------------------------------
// Core 32x64 GEMM strip per wave: acc[0..3] = rows m0..m0+15, acc[4..7] = +16.
// A: [M][K] f16 row-major.  Wt: [N][K] f16 (transposed weight).
// Pure b128 loads + 8 wmma per 32-K step; no conversions.
// ---------------------------------------------------------------------------
__device__ __forceinline__ void gemm32x64(
    const half_t* __restrict__ A, const half_t* __restrict__ Wt,
    int m0, int n0, int lane, v8f (&acc)[8]) {
  const int col = lane & 15;
  const int kb  = (lane >> 4) * 8;
  const half_t* arow0 = A + (size_t)(m0 + col) * DMODEL;
  const half_t* arow1 = arow0 + (size_t)16 * DMODEL;
  for (int ks = 0; ks < DMODEL; ks += 32) {
    frag16 a0, a1;
    a0.h[0] = *(const v8h*)(arow0 + ks + kb);
    a0.h[1] = *(const v8h*)(arow0 + ks + kb + 16);
    a1.h[0] = *(const v8h*)(arow1 + ks + kb);
    a1.h[1] = *(const v8h*)(arow1 + ks + kb + 16);
    const int kbb = ks + (lane >> 4) * 8;
#pragma unroll
    for (int t = 0; t < 4; ++t) {
      frag16 b;
      const half_t* wrow = Wt + (size_t)(n0 + t * 16 + col) * DMODEL + kbb;
      b.h[0] = *(const v8h*)(wrow);
      b.h[1] = *(const v8h*)(wrow + 16);
      acc[t]     = wmma_f16(a0.v, b.v, acc[t]);      // B reused across 2 M-tiles
      acc[4 + t] = wmma_f16(a1.v, b.v, acc[4 + t]);
    }
  }
}

// ---------------------------------------------------------------------------
// Kernel 1: fused QKV projection, f16 out scattered to [B,H,S,Dh]
// ---------------------------------------------------------------------------
__global__ __launch_bounds__(256) void qkv_proj_wmma(
    const half_t* __restrict__ xh, const half_t* __restrict__ wt,
    half_t* __restrict__ qkv) {
  const int lane = threadIdx.x & 31;
  const int wave = threadIdx.x >> 5;
  const int tile = blockIdx.x * 8 + wave;   // 4096 tiles = 256 Mtiles * 16 strips
  const int m0 = (tile & 255) * 32;
  const int n0 = (tile >> 8) * 64;
  const half_t* Wt = wt + (size_t)blockIdx.z * DMODEL * DMODEL;
  half_t* outbase = qkv + (size_t)blockIdx.z * ((size_t)BSROWS * DMODEL);

  v8f acc[8] = {};
  gemm32x64(xh, Wt, m0, n0, lane, acc);

  const int col = lane & 15;
#pragma unroll
  for (int g = 0; g < 2; ++g) {
#pragma unroll
    for (int r = 0; r < 8; ++r) {
      const int bs = m0 + g * 16 + r + ((lane >> 4) << 3);
      const int b  = bs >> 12;
      const int s  = bs & (SEQ - 1);
#pragma unroll
      for (int t = 0; t < 4; ++t) {
        const int d = n0 + t * 16 + col;
        outbase[((size_t)(b * HEADS + (d >> 6)) * SEQ + s) * DHEAD + (d & 63)] =
            (half_t)acc[g * 4 + t][r];
      }
    }
  }
}

// ---------------------------------------------------------------------------
// Kernel 2: flash attention (online softmax, 32-key tiles), f16 output [B,S,D]
// ---------------------------------------------------------------------------
__global__ __launch_bounds__(256) void flash_attn_wmma(
    const half_t* __restrict__ q, const half_t* __restrict__ k,
    const half_t* __restrict__ v, half_t* __restrict__ out) {
  __shared__ half_t lds[8 * (32 * 64 + 16 * 32)];
  const int lane = threadIdx.x & 31;
  const int wave = threadIdx.x >> 5;
  half_t* ldsV = lds + wave * (32 * 64 + 16 * 32);
  half_t* ldsP = ldsV + 32 * 64;

  const int gw = blockIdx.x * 8 + wave;     // 8192 waves
  const int qt = gw & 255;
  const int bh = gw >> 8;
  const half_t* qbase = q + (size_t)bh * SEQ * DHEAD;
  const half_t* kbase = k + (size_t)bh * SEQ * DHEAD;
  const half_t* vbase = v + (size_t)bh * SEQ * DHEAD;

  const int col = lane & 15;
  const int kb  = (lane >> 4) * 8;

  frag16 qa[2];
  {
    const half_t* qrow = qbase + (size_t)(qt * 16 + col) * DHEAD;
#pragma unroll
    for (int c = 0; c < 2; ++c) {
      qa[c].h[0] = *(const v8h*)(qrow + c * 32 + kb);
      qa[c].h[1] = *(const v8h*)(qrow + c * 32 + kb + 16);
    }
  }

  float mrow[8], lrow[8];
  v8f o[4] = {};
#pragma unroll
  for (int r = 0; r < 8; ++r) { mrow[r] = -INFINITY; lrow[r] = 0.0f; }
  const float scale = 0.125f;

  for (int kv = 0; kv < SEQ; kv += 32) {
    if (kv + 32 < SEQ) {
      __builtin_prefetch(kbase + (size_t)(kv + 32 + lane) * DHEAD, 0, 3);
      __builtin_prefetch(vbase + (size_t)(kv + 32 + lane) * DHEAD, 0, 3);
    }

    // ---- stage V tile (32x64 f16) into LDS: async DMA if available
    {
      const half_t* vrow = vbase + (size_t)(kv + lane) * DHEAD;
#if USE_ASYNC_LDS
#pragma unroll
      for (int i = 0; i < 8; ++i)
        __builtin_amdgcn_global_load_async_to_lds_b128(
            (gptr_v4i)(vrow + i * 8),
            (lptr_v4i)(ldsV + lane * 64 + i * 8),
            0, 0);
#else
#pragma unroll
      for (int i = 0; i < 8; ++i)
        *(v8h*)(ldsV + lane * 64 + i * 8) = *(const v8h*)(vrow + i * 8);
#endif
    }

    // ---- S = Q K^T : two 16-key sub-tiles, Dh split 0-31 / 32-63
    v8f sacc[2] = {};
#pragma unroll
    for (int ksub = 0; ksub < 2; ++ksub) {
      const half_t* krow = kbase + (size_t)(kv + ksub * 16 + col) * DHEAD;
#pragma unroll
      for (int c = 0; c < 2; ++c) {
        frag16 b;
        b.h[0] = *(const v8h*)(krow + c * 32 + (lane >> 4) * 8);
        b.h[1] = *(const v8h*)(krow + c * 32 + (lane >> 4) * 8 + 16);
        sacc[ksub] = wmma_f16(qa[c].v, b.v, sacc[ksub]);
      }
    }

    // ---- online softmax; row = r + 8*(lane>=16), col = lane&15
#pragma unroll
    for (int r = 0; r < 8; ++r) {
      float s0 = sacc[0][r] * scale;
      float s1 = sacc[1][r] * scale;
      float mx = fmaxf(s0, s1);
#pragma unroll
      for (int off = 1; off < 16; off <<= 1)
        mx = fmaxf(mx, __shfl_xor(mx, off, 32));
      const float mnew  = fmaxf(mrow[r], mx);
      const float alpha = __expf(mrow[r] - mnew);
      const float p0 = __expf(s0 - mnew);
      const float p1 = __expf(s1 - mnew);
      float ps = p0 + p1;
#pragma unroll
      for (int off = 1; off < 16; off <<= 1)
        ps += __shfl_xor(ps, off, 32);
      lrow[r] = lrow[r] * alpha + ps;
      mrow[r] = mnew;
      const int prow = r + ((lane >> 4) << 3);
      ldsP[prow * 32 + col]      = (half_t)p0;   // transpose P via LDS
      ldsP[prow * 32 + 16 + col] = (half_t)p1;
#pragma unroll
      for (int t = 0; t < 4; ++t) o[t][r] *= alpha;
    }
#if USE_ASYNC_LDS
    __builtin_amdgcn_s_wait_asynccnt(0);        // V tile landed in LDS
#endif
    __builtin_amdgcn_wave_barrier();

    // ---- P as 16x32 A-fragment from LDS
    frag16 pa;
    pa.h[0] = *(const v8h*)(ldsP + col * 32 + kb);
    pa.h[1] = *(const v8h*)(ldsP + col * 32 + kb + 16);

    // ---- O += P @ V : 4 Dh column tiles
    const int krow0 = (lane >> 4) * 8;
#pragma unroll
    for (int t = 0; t < 4; ++t) {
      frag16 b;
      const int vc = t * 16 + col;
#pragma unroll
      for (int i = 0; i < 16; ++i) b.v[i] = ldsV[(krow0 + i + (i & 8)) * 64 + vc];
      o[t] = wmma_f16(pa.v, b.v, o[t]);
    }
    __builtin_amdgcn_wave_barrier();
  }

  // ---- epilogue: O / l -> f16 attn [B,S,1024]
  const int b = bh >> 4, h = bh & 15;
#pragma unroll
  for (int r = 0; r < 8; ++r) {
    const int m = r + ((lane >> 4) << 3);
    const int s = qt * 16 + m;
    const float inv = 1.0f / lrow[r];
#pragma unroll
    for (int t = 0; t < 4; ++t)
      out[((size_t)b * SEQ + s) * DMODEL + h * DHEAD + t * 16 + col] =
          (half_t)(o[t][r] * inv);
  }
}

// ---------------------------------------------------------------------------
// Kernel 3: output projection -> f32 d_out
// ---------------------------------------------------------------------------
__global__ __launch_bounds__(256) void out_proj_wmma(
    const half_t* __restrict__ attn, const half_t* __restrict__ wt_o,
    float* __restrict__ out) {
  const int lane = threadIdx.x & 31;
  const int wave = threadIdx.x >> 5;
  const int tile = blockIdx.x * 8 + wave;
  const int m0 = (tile & 255) * 32;
  const int n0 = (tile >> 8) * 64;

  v8f acc[8] = {};
  gemm32x64(attn, wt_o, m0, n0, lane, acc);

  const int col = lane & 15;
#pragma unroll
  for (int g = 0; g < 2; ++g)
#pragma unroll
    for (int r = 0; r < 8; ++r) {
      const int m = m0 + g * 16 + r + ((lane >> 4) << 3);
#pragma unroll
      for (int t = 0; t < 4; ++t)
        out[(size_t)m * DMODEL + n0 + t * 16 + col] = acc[g * 4 + t][r];
    }
}

// ---------------------------------------------------------------------------
extern "C" void kernel_launch(void* const* d_in, const int* in_sizes, int n_in,
                              void* d_out, int out_size, void* d_ws, size_t ws_size,
                              hipStream_t stream) {
  const float* x  = (const float*)d_in[0];
  const float* wq = (const float*)d_in[1];
  const float* wk = (const float*)d_in[2];
  const float* wv = (const float*)d_in[3];
  const float* wo = (const float*)d_in[4];

  const size_t x_elems = (size_t)BSROWS * DMODEL;   // 8M
  const size_t w_elems = (size_t)DMODEL * DMODEL;   // 1M

  char* ws = (char*)d_ws;
  half_t* xh   = (half_t*)ws;                               ws += x_elems * 2;     // 16 MB
  half_t* wt   = (half_t*)ws;                               ws += 4 * w_elems * 2; //  8 MB
  half_t* qkv  = (half_t*)ws;                               ws += 3 * x_elems * 2; // 48 MB
  half_t* attn = (half_t*)ws;                                                      // 16 MB

  // 1) repack inputs to f16 (x) and f16-transposed (weights)
  cvt_f32_to_f16<<<x_elems / (256 * 4), 256, 0, stream>>>(x, xh);
  cvt_transpose_w<<<dim3(32, 32, 4), 256, 0, stream>>>(wq, wk, wv, wo, wt);

  // 2) Q,K,V projections (z selects weight), f16 out in [B,H,S,Dh]
  qkv_proj_wmma<<<dim3(512, 1, 3), 256, 0, stream>>>(xh, wt, qkv);

  // 3) flash attention -> f16 attn [B,S,1024]
  flash_attn_wmma<<<1024, 256, 0, stream>>>(
      qkv, qkv + x_elems, qkv + 2 * x_elems, attn);

  // 4) output projection (Wt index 3 = wo) -> f32 d_out
  out_proj_wmma<<<512, 256, 0, stream>>>(attn, wt + 3 * w_elems, (float*)d_out);
}
